// MultiHeadAttention_13898514170452
// MI455X (gfx1250) — matrix operand order repaired and checked by reading, not compiled, not run on hardware
//
#include <hip/hip_runtime.h>
#include <hip/hip_bf16.h>

typedef __attribute__((ext_vector_type(16))) _Float16 v16h;
typedef __attribute__((ext_vector_type(8)))  _Float16 v8h;
typedef __attribute__((ext_vector_type(8)))  float    v8f;

#define WMMA_F16(a, b, c) \
  __builtin_amdgcn_wmma_f32_16x16x32_f16(false, (a), false, (b), (short)0, (c), false, false)

static constexpr int Bn = 4, Tn = 2048, Cn = 1024, Hn = 16, DHn = 64;

union HV16 {
  v16h v;
  v8h  h[2];
};

// A-fragment (16x32 f16): row-major source, K contiguous per lane half.
__device__ __forceinline__ v16h load_afrag(const _Float16* __restrict__ rowk0,
                                           int laneHi) {
  HV16 u;
  u.h[0] = *(const v8h*)(rowk0 + 16 * laneHi);
  u.h[1] = *(const v8h*)(rowk0 + 16 * laneHi + 8);
  return u.v;
}

// Fully-unrolled K-loop, 32x64 tile per wave (2 A row-groups share B-frags),
// 2-deep ping-pong prefetch. 8 WMMAs per chunk vs 12 b128 loads.
template <int NCH>
__device__ __forceinline__ void gemm_pipeline2(const _Float16* __restrict__ Arow0,
                                               const _Float16* __restrict__ Arow1,
                                               const _Float16* const* Bptr,
                                               int laneHi, v8f acc[2][4]) {
  v16h af[2][2];
  v16h bf[2][4];
  af[0][0] = load_afrag(Arow0, laneHi);
  af[0][1] = load_afrag(Arow1, laneHi);
#pragma unroll
  for (int n = 0; n < 4; ++n) bf[0][n] = *(const v16h*)(Bptr[n]);
#pragma unroll
  for (int i = 0; i < NCH; ++i) {
    const int cur = i & 1, nxt = cur ^ 1;
    if (i + 1 < NCH) {
      const int koff = (i + 1) * 32;
      af[nxt][0] = load_afrag(Arow0 + koff, laneHi);
      af[nxt][1] = load_afrag(Arow1 + koff, laneHi);
#pragma unroll
      for (int n = 0; n < 4; ++n) bf[nxt][n] = *(const v16h*)(Bptr[n] + koff);
    }
#pragma unroll
    for (int n = 0; n < 4; ++n) {
      acc[0][n] = WMMA_F16(af[cur][0], bf[cur][n], acc[0][n]);
      acc[1][n] = WMMA_F16(af[cur][1], bf[cur][n], acc[1][n]);
    }
  }
}

// ---------------------------------------------------------------------------
// Kernel 1: plain f32 -> f16 conversion
// ---------------------------------------------------------------------------
__global__ void cvt_f32_f16(const float* __restrict__ src,
                            _Float16* __restrict__ dst, int n) {
  int i = blockIdx.x * blockDim.x + threadIdx.x;
  if (i < n) dst[i] = (_Float16)src[i];
}

// ---------------------------------------------------------------------------
// Kernel 2: f32 -> f16 with per-slab transpose: src[slab][R][Cc] -> dst[slab][Cc][R]
// ---------------------------------------------------------------------------
__global__ void cvt_transpose_f16(const float* __restrict__ src,
                                  _Float16* __restrict__ dst, int R, int Cc) {
  int i = blockIdx.x * blockDim.x + threadIdx.x;
  size_t slab = (size_t)blockIdx.y * R * Cc;
  if (i < R * Cc) {
    int r = i / Cc, c = i % Cc;
    dst[slab + (size_t)c * R + r] = (_Float16)src[slab + (size_t)r * Cc + c];
  }
}

// ---------------------------------------------------------------------------
// Kernel 3: per-head projection (LDS-free, pipelined, 32x64 tile per wave)
//   X16[b]: [T,C] f16   WT[h]: [DH,C] f16 (pre-transposed)
//   out: transposed==0 -> [B,H,T,DH] ; transposed==1 -> [B,H,DH,T]
// grid: (T/128, B*H), block: 128 (4 waves); each wave: 32x64 strip
// ---------------------------------------------------------------------------
__global__ __launch_bounds__(128)
void proj_kernel(const _Float16* __restrict__ X16,
                 const _Float16* __restrict__ WT,
                 _Float16* __restrict__ Y, int transposed) {
  const int tid = threadIdx.x;
  const int lane = tid & 31, wv = tid >> 5;
  const int laneHi = lane >> 4, laneLo = lane & 15;
  const int m0 = blockIdx.x * 128;
  const int bh = blockIdx.y;
  const int b = bh / Hn, h = bh % Hn;
  const int mw = m0 + wv * 32;

  const _Float16* Arow0 = X16 + ((size_t)b * Tn + (mw + laneLo)) * Cn;
  const _Float16* Arow1 = X16 + ((size_t)b * Tn + (mw + 16 + laneLo)) * Cn;
  const _Float16* Bptr[4];
#pragma unroll
  for (int n = 0; n < 4; ++n)
    Bptr[n] = WT + ((size_t)h * DHn + (n * 16 + laneLo)) * Cn + 16 * laneHi;

  v8f acc[2][4] = {};
  gemm_pipeline2<Cn / 32>(Arow0, Arow1, Bptr, laneHi, acc);

  if (transposed) {  // Y[b,h,d,t]  (for V)
    _Float16* Yp = Y + (size_t)bh * DHn * Tn;
#pragma unroll
    for (int g = 0; g < 2; ++g)
#pragma unroll
      for (int n = 0; n < 4; ++n)
#pragma unroll
        for (int r = 0; r < 8; ++r) {
          int t = mw + g * 16 + r + laneHi * 8;
          int d = n * 16 + laneLo;
          Yp[(size_t)d * Tn + t] = (_Float16)acc[g][n][r];
        }
  } else {           // Y[b,h,t,d]  (for Q, K)
    _Float16* Yp = Y + (size_t)bh * Tn * DHn;
#pragma unroll
    for (int g = 0; g < 2; ++g)
#pragma unroll
      for (int n = 0; n < 4; ++n)
#pragma unroll
        for (int r = 0; r < 8; ++r) {
          int t = mw + g * 16 + r + laneHi * 8;
          int d = n * 16 + laneLo;
          Yp[(size_t)t * DHn + d] = (_Float16)acc[g][n][r];
        }
  }
}

// ---------------------------------------------------------------------------
// Kernel 4: causal flash attention (only LDS use: per-wave P relayout)
//   Q,K : [B,H,T,DH] f16     VT : [B,H,DH,T] f16
//   O   : [B,T,H*DH] f16
// grid: (T/64, B*H), block: 128 (4 waves); each wave: 16 query rows
// ---------------------------------------------------------------------------
__global__ __launch_bounds__(128)
void attn_kernel(const _Float16* __restrict__ Q,
                 const _Float16* __restrict__ K,
                 const _Float16* __restrict__ VT,
                 _Float16* __restrict__ O) {
  __shared__ _Float16 pS[4][16][80];  // per-wave P buffer (C->A relayout)
  const int tid = threadIdx.x;
  const int lane = tid & 31, wv = tid >> 5;
  const int laneHi = lane >> 4, laneLo = lane & 15;
  const int qb = blockIdx.x;
  const int bh = blockIdx.y;
  const int b = bh >> 4, h = bh & 15;
  const int qbase = qb * 64;

  const _Float16* Qp  = Q  + (size_t)bh * Tn * DHn;
  const _Float16* Kp  = K  + (size_t)bh * Tn * DHn;
  const _Float16* VTp = VT + (size_t)bh * DHn * Tn;

  // Q A-fragments, resident for the whole pass
  v16h qf[2];
  {
    const _Float16* qrow = Qp + (size_t)(qbase + wv * 16 + laneLo) * DHn;
    qf[0] = load_afrag(qrow, laneHi);
    qf[1] = load_afrag(qrow + 32, laneHi);
  }
  const _Float16* KrowBase[4];
#pragma unroll
  for (int n = 0; n < 4; ++n)
    KrowBase[n] = Kp + (size_t)(n * 16 + laneLo) * DHn + 16 * laneHi;
  const _Float16* VTrow[4];
#pragma unroll
  for (int n = 0; n < 4; ++n)
    VTrow[n] = VTp + (size_t)(n * 16 + laneLo) * Tn + 16 * laneHi;

  v8f o[4] = {};
  float mrow[8], lrow[8];
#pragma unroll
  for (int r = 0; r < 8; ++r) { mrow[r] = -1e30f; lrow[r] = 0.f; }

  for (int sb = 0; sb <= qb; ++sb) {
    const int sbase = sb * 64;

    // batched fragment loads: K first (needed first), then V (consumed after
    // softmax -> its latency hides behind the VALU-heavy section)
    v16h kf[4][2], vf[4][2];
#pragma unroll
    for (int n = 0; n < 4; ++n)
#pragma unroll
      for (int c = 0; c < 2; ++c)
        kf[n][c] = *(const v16h*)(KrowBase[n] + (size_t)sbase * DHn + c * 32);
#pragma unroll
    for (int n = 0; n < 4; ++n)
#pragma unroll
      for (int c = 0; c < 2; ++c)
        vf[n][c] = *(const v16h*)(VTrow[n] + sbase + c * 32);

    if (sb < qb) {  // prefetch next s-block (uniform branch)
      const int nb = sbase + 64;
      __builtin_prefetch(Kp + (size_t)(nb + lane) * DHn, 0, 3);
      __builtin_prefetch(Kp + (size_t)(nb + 32 + lane) * DHn, 0, 3);
      __builtin_prefetch(VTp + (size_t)(lane * 2) * Tn + nb, 0, 3);
    }

    // S = Q * K^T : 8 WMMAs
    v8f s[4];
#pragma unroll
    for (int n = 0; n < 4; ++n) {
      v8f z = {};
      z = WMMA_F16(qf[0], kf[n][0], z);
      s[n] = WMMA_F16(qf[1], kf[n][1], z);
    }

    // online softmax
    const bool diag = (sb == qb);
    float pnew[4][8];
#pragma unroll
    for (int r = 0; r < 8; ++r) {
      int tg = qbase + wv * 16 + r + laneHi * 8;
      float rm = -1e30f;
#pragma unroll
      for (int n = 0; n < 4; ++n) {
        float val = s[n][r] * 0.125f;  // 1/sqrt(64)
        int sg = sbase + n * 16 + laneLo;
        if (diag && sg > tg) val = -1e30f;  // causal mask
        pnew[n][r] = val;
        rm = fmaxf(rm, val);
      }
#pragma unroll
      for (int off = 1; off < 16; off <<= 1)
        rm = fmaxf(rm, __shfl_xor(rm, off));
      float mold = mrow[r];
      float mnew = fmaxf(mold, rm);
      float corr = __expf(mold - mnew);
      float rs = 0.f;
#pragma unroll
      for (int n = 0; n < 4; ++n) {
        float p = __expf(pnew[n][r] - mnew);
        pnew[n][r] = p;
        rs += p;
      }
#pragma unroll
      for (int off = 1; off < 16; off <<= 1)
        rs += __shfl_xor(rs, off);
      lrow[r] = lrow[r] * corr + rs;
      mrow[r] = mnew;
#pragma unroll
      for (int n = 0; n < 4; ++n) o[n][r] *= corr;
    }

    // P: C-layout -> A-layout via per-wave LDS (reads vectorized)
#pragma unroll
    for (int n = 0; n < 4; ++n)
#pragma unroll
      for (int r = 0; r < 8; ++r)
        pS[wv][r + laneHi * 8][n * 16 + laneLo] = (_Float16)pnew[n][r];
    asm volatile("s_wait_dscnt 0" ::: "memory");
    v16h pf[2];
#pragma unroll
    for (int c = 0; c < 2; ++c) {
      HV16 u;
      u.h[0] = *(const v8h*)&pS[wv][laneLo][c * 32 + 16 * laneHi];
      u.h[1] = *(const v8h*)&pS[wv][laneLo][c * 32 + 16 * laneHi + 8];
      pf[c] = u.v;
    }

    // O += P * V : 8 WMMAs (V already resident)
#pragma unroll
    for (int n = 0; n < 4; ++n) {
      o[n] = WMMA_F16(pf[0], vf[n][0], o[n]);
      o[n] = WMMA_F16(pf[1], vf[n][1], o[n]);
    }
  }

  // epilogue: O / l, concat-head layout [B,T,H*DH]
#pragma unroll
  for (int n = 0; n < 4; ++n)
#pragma unroll
    for (int r = 0; r < 8; ++r) {
      int t = qbase + wv * 16 + r + laneHi * 8;
      int d = n * 16 + laneLo;
      O[((size_t)b * Tn + t) * (Hn * DHn) + h * DHn + d] =
          (_Float16)(o[n][r] / lrow[r]);
    }
}

// ---------------------------------------------------------------------------
// Kernel 5: output projection (LDS-free, pipelined, 32x64 tile per wave)
//   A[8192,1024] f16 x WoT[1024(C),1024(HD)] f16 + bo -> f32
// grid: (C/64, B*T/128), block: 128
// ---------------------------------------------------------------------------
__global__ __launch_bounds__(128)
void outproj_kernel(const _Float16* __restrict__ A,
                    const _Float16* __restrict__ WoT,
                    const float* __restrict__ bias,
                    float* __restrict__ Y) {
  const int tid = threadIdx.x;
  const int lane = tid & 31, wv = tid >> 5;
  const int laneHi = lane >> 4, laneLo = lane & 15;
  const int n0 = blockIdx.x * 64;
  const int m0 = blockIdx.y * 128;
  const int Kd = Hn * DHn;  // 1024
  const int mw = m0 + wv * 32;

  const _Float16* Arow0 = A + (size_t)(mw + laneLo) * Kd;
  const _Float16* Arow1 = A + (size_t)(mw + 16 + laneLo) * Kd;
  const _Float16* Bptr[4];
#pragma unroll
  for (int n = 0; n < 4; ++n)
    Bptr[n] = WoT + (size_t)(n0 + n * 16 + laneLo) * Kd + 16 * laneHi;

  v8f acc[2][4] = {};
  gemm_pipeline2<(Hn * DHn) / 32>(Arow0, Arow1, Bptr, laneHi, acc);

#pragma unroll
  for (int g = 0; g < 2; ++g)
#pragma unroll
    for (int n = 0; n < 4; ++n)
#pragma unroll
      for (int r = 0; r < 8; ++r) {
        int row = mw + g * 16 + r + laneHi * 8;
        int col = n0 + n * 16 + laneLo;
        Y[(size_t)row * Cn + col] = acc[g][n][r] + bias[col];
      }
}

// ---------------------------------------------------------------------------
extern "C" void kernel_launch(void* const* d_in, const int* in_sizes, int n_in,
                              void* d_out, int out_size, void* d_ws, size_t ws_size,
                              hipStream_t stream) {
  const float* k  = (const float*)d_in[0];
  const float* q  = (const float*)d_in[1];
  const float* v  = (const float*)d_in[2];
  const float* Wk = (const float*)d_in[3];
  const float* Wq = (const float*)d_in[4];
  const float* Wv = (const float*)d_in[5];
  const float* Wo = (const float*)d_in[6];
  const float* bo = (const float*)d_in[7];
  float* out = (float*)d_out;

  const size_t actElems = (size_t)Bn * Tn * Cn;         // 8388608 (= B*H*T*DH)
  const size_t wElems   = (size_t)Hn * Cn * DHn;        // 1048576

  _Float16* ws   = (_Float16*)d_ws;
  _Float16* q16  = ws;                   // [B,T,C]
  _Float16* k16  = q16 + actElems;
  _Float16* v16  = k16 + actElems;
  _Float16* qh   = v16 + actElems;       // [B,H,T,DH]
  _Float16* kh   = qh + actElems;
  _Float16* vhT  = kh + actElems;        // [B,H,DH,T]
  _Float16* att  = vhT + actElems;       // [B,T,H*DH]
  _Float16* wqT  = att + actElems;       // [H,DH,C]
  _Float16* wkT  = wqT + wElems;
  _Float16* wvT  = wkT + wElems;
  _Float16* woT  = wvT + wElems;         // [C, H*DH]

  const int blk = 256;
  cvt_f32_f16<<<(int)((actElems + blk - 1) / blk), blk, 0, stream>>>(q, q16, (int)actElems);
  cvt_f32_f16<<<(int)((actElems + blk - 1) / blk), blk, 0, stream>>>(k, k16, (int)actElems);
  cvt_f32_f16<<<(int)((actElems + blk - 1) / blk), blk, 0, stream>>>(v, v16, (int)actElems);
  {
    int per = Cn * DHn;
    dim3 g((per + blk - 1) / blk, Hn);
    cvt_transpose_f16<<<g, blk, 0, stream>>>(Wq, wqT, Cn, DHn);
    cvt_transpose_f16<<<g, blk, 0, stream>>>(Wk, wkT, Cn, DHn);
    cvt_transpose_f16<<<g, blk, 0, stream>>>(Wv, wvT, Cn, DHn);
    dim3 go((Cn * Hn * DHn + blk - 1) / blk, 1);
    cvt_transpose_f16<<<go, blk, 0, stream>>>(Wo, woT, Hn * DHn, Cn);
  }

  dim3 pg(Tn / 128, Bn * Hn);
  proj_kernel<<<pg, 128, 0, stream>>>(q16, wqT, qh, 0);
  proj_kernel<<<pg, 128, 0, stream>>>(k16, wkT, kh, 0);
  proj_kernel<<<pg, 128, 0, stream>>>(v16, wvT, vhT, 1);

  attn_kernel<<<dim3(Tn / 64, Bn * Hn), 128, 0, stream>>>(qh, kh, vhT, att);

  outproj_kernel<<<dim3(Cn / 64, (Bn * Tn) / 128), 128, 0, stream>>>(att, woT, bo, out);
}
